// VectorQuantizer1_1494648619095
// MI455X (gfx1250) — matrix operand order repaired and checked by reading, not compiled, not run on hardware
//
#include <hip/hip_runtime.h>
#include <hip/hip_bf16.h>

// VQ-VAE quantizer for MI455X (gfx1250, wave32, WMMA).
// dist(n,k) = ||w_k||^2 - 2 * x_n . w_k   (||x_n||^2 constant per row, dropped)
// GEMM runs on v_wmma_f32_16x16x32_bf16 with a hi/lo bf16 split of the fp32
// inputs (3 WMMAs per k-step, lo*lo dropped) so argmin ranking stays
// ~fp32-accurate. w is pre-split into bf16 hi/lo arrays once (same bytes as
// fp32) so the hot loop does zero conversion VALU work.

typedef __attribute__((ext_vector_type(16))) __bf16 v16bf;
typedef __attribute__((ext_vector_type(8)))  float  v8f;

#define DIMS     512
#define KSTEPS   16      // 512 / 32
#define NCODES   8192
#define NTOK     16384
#define NSPLIT   4       // code-range split across wave columns

// ------------------------------------------- w -> hi/lo bf16 + wnorm prep --
__global__ __launch_bounds__(256) void prep_w_kernel(const float* __restrict__ w,
                                                     __bf16* __restrict__ w_hi,
                                                     __bf16* __restrict__ w_lo,
                                                     float* __restrict__ wnorm) {
  const int wave = threadIdx.x >> 5, lane = threadIdx.x & 31;
  const int row = blockIdx.x * 8 + wave;          // grid = 1024 -> 8192 rows
  const float* wr = w + (size_t)row * DIMS;
  __bf16* hr = w_hi + (size_t)row * DIMS;
  __bf16* lr = w_lo + (size_t)row * DIMS;
  float s = 0.0f;
  #pragma unroll
  for (int i = 0; i < DIMS / 32; ++i) {
    const int e = lane + i * 32;
    const float f = wr[e];
    s += f * f;
    const __bf16 h = (__bf16)f;
    hr[e] = h;
    lr[e] = (__bf16)(f - (float)h);
  }
  #pragma unroll
  for (int off = 16; off; off >>= 1) s += __shfl_xor(s, off, 32);
  if (lane == 0) wnorm[row] = s;
}

// ------------------------------------------------------- fused GEMM+argmin --
__global__ __launch_bounds__(256) void vq_argmin_kernel(const float* __restrict__ x,
                                                        const __bf16* __restrict__ w_hi,
                                                        const __bf16* __restrict__ w_lo,
                                                        const float* __restrict__ wnorm,
                                                        float* __restrict__ cand_val,
                                                        int*   __restrict__ cand_idx) {
  // A (x tile, 32 tokens x 512 dims) staged in LDS as hi/lo bf16, already in
  // per-lane WMMA A-fragment order: [wm][kstep][lane][16 bf16].
  __shared__ __align__(32) __bf16 a_hi[2][KSTEPS][32][16];
  __shared__ __align__(32) __bf16 a_lo[2][KSTEPS][32][16];

  const int tid = threadIdx.x;
  const int blk = blockIdx.x;                     // grid = 512 -> 32 tokens each

  // ---- stage A fragments (cooperative, 1024 (wm,ks,lane) triples / 256 thr)
  for (int idx = tid; idx < 2 * KSTEPS * 32; idx += 256) {
    const int wm   = idx >> 9;          // / 512
    const int rem  = idx & 511;
    const int ks   = rem >> 5;
    const int lane = rem & 31;
    const int row  = blk * 32 + wm * 16 + (lane & 15);
    // A order used here (contiguous, B matches): lanes 0-15 hold K {0..15}, lanes 16-31 hold {16..31}
    const float* xr = x + (size_t)row * DIMS + ks * 32 + ((lane & 16) ? 16 : 0);
    #pragma unroll
    for (int j = 0; j < 8; ++j) {
      const int k0 = 2 * j;
      const float f0 = xr[k0], f1 = xr[k0 + 1];
      const __bf16 h0 = (__bf16)f0, h1 = (__bf16)f1;
      a_hi[wm][ks][lane][2 * j]     = h0;
      a_hi[wm][ks][lane][2 * j + 1] = h1;
      a_lo[wm][ks][lane][2 * j]     = (__bf16)(f0 - (float)h0);
      a_lo[wm][ks][lane][2 * j + 1] = (__bf16)(f1 - (float)h1);
    }
  }
  __syncthreads();

  const int wave  = tid >> 5;
  const int lane  = tid & 31;
  const int wm    = wave & 1;                     // which 16-token slice
  const int wn    = wave >> 1;                    // which code quarter
  const int nlane = lane & 15;
  const int khalf = (lane & 16) ? 16 : 0;         // B order used here (matches A): hi lanes hold K+16..31
  const int nbase0 = wn * (NCODES / NSPLIT);

  float minv0[8], minv1[8];
  int   mini0[8], mini1[8];
  #pragma unroll
  for (int j = 0; j < 8; ++j) {
    minv0[j] = 3.4e38f; mini0[j] = 0;
    minv1[j] = 3.4e38f; mini1[j] = 0;
  }

  for (int chunk = 0; chunk < (NCODES / NSPLIT) / 32; ++chunk) {
    const int n0 = nbase0 + chunk * 32 + nlane;   // two code columns per lane
    const int n1 = n0 + 16;
    const __bf16* b0h = w_hi + (size_t)n0 * DIMS + khalf;
    const __bf16* b0l = w_lo + (size_t)n0 * DIMS + khalf;
    const __bf16* b1h = w_hi + (size_t)n1 * DIMS + khalf;
    const __bf16* b1l = w_lo + (size_t)n1 * DIMS + khalf;
    v8f acc0 = {}, acc1 = {};
    #pragma unroll 2
    for (int ks = 0; ks < KSTEPS; ++ks) {
      const v16bf ah = *(const v16bf*)&a_hi[wm][ks][lane][0];
      const v16bf al = *(const v16bf*)&a_lo[wm][ks][lane][0];
      const v16bf bh0 = *(const v16bf*)(b0h + ks * 32);
      const v16bf bl0 = *(const v16bf*)(b0l + ks * 32);
      const v16bf bh1 = *(const v16bf*)(b1h + ks * 32);
      const v16bf bl1 = *(const v16bf*)(b1l + ks * 32);
      // dot += ah*bh + al*bh + ah*bl  (drop lo*lo); A frags reused for 2 cols
      acc0 = __builtin_amdgcn_wmma_f32_16x16x32_bf16(false, ah, false, bh0,
                                                     (short)0, acc0, false, false);
      acc1 = __builtin_amdgcn_wmma_f32_16x16x32_bf16(false, ah, false, bh1,
                                                     (short)0, acc1, false, false);
      acc0 = __builtin_amdgcn_wmma_f32_16x16x32_bf16(false, al, false, bh0,
                                                     (short)0, acc0, false, false);
      acc1 = __builtin_amdgcn_wmma_f32_16x16x32_bf16(false, al, false, bh1,
                                                     (short)0, acc1, false, false);
      acc0 = __builtin_amdgcn_wmma_f32_16x16x32_bf16(false, ah, false, bl0,
                                                     (short)0, acc0, false, false);
      acc1 = __builtin_amdgcn_wmma_f32_16x16x32_bf16(false, ah, false, bl1,
                                                     (short)0, acc1, false, false);
    }
    const float wn0 = wnorm[n0];
    const float wn1 = wnorm[n1];
    #pragma unroll
    for (int j = 0; j < 8; ++j) {                 // C: lane=N, vgpr j -> M row
      const float d0 = wn0 - 2.0f * acc0[j];
      const float d1 = wn1 - 2.0f * acc1[j];
      if (d0 < minv0[j]) { minv0[j] = d0; mini0[j] = n0; }  // strict < keeps lowest idx
      if (d1 < minv1[j]) { minv1[j] = d1; mini1[j] = n1; }
    }
  }

  // merge the two column streams (tie -> lower index)
  #pragma unroll
  for (int j = 0; j < 8; ++j) {
    if (minv1[j] < minv0[j] || (minv1[j] == minv0[j] && mini1[j] < mini0[j])) {
      minv0[j] = minv1[j]; mini0[j] = mini1[j];
    }
  }

  // min-reduce across the 16 lanes sharing each M row (lanes 0-15 / 16-31)
  #pragma unroll
  for (int off = 1; off < 16; off <<= 1) {
    #pragma unroll
    for (int j = 0; j < 8; ++j) {
      const float ov = __shfl_xor(minv0[j], off, 32);
      const int   oi = __shfl_xor(mini0[j], off, 32);
      if (ov < minv0[j] || (ov == minv0[j] && oi < mini0[j])) { minv0[j] = ov; mini0[j] = oi; }
    }
  }
  if (nlane == 0) {
    const int mbase = blk * 32 + wm * 16 + ((lane & 16) ? 8 : 0);
    #pragma unroll
    for (int j = 0; j < 8; ++j) {
      const int tok = mbase + j;
      cand_val[tok * NSPLIT + wn] = minv0[j];
      cand_idx[tok * NSPLIT + wn] = mini0[j];
    }
  }
}

// ------------------------------------------------- combine split candidates --
__global__ __launch_bounds__(256) void combine_kernel(const float* __restrict__ cand_val,
                                                      const int* __restrict__ cand_idx,
                                                      int* __restrict__ ids) {
  const int t = blockIdx.x * 256 + threadIdx.x;
  if (t >= NTOK) return;
  float bv = cand_val[t * NSPLIT];
  int   bi = cand_idx[t * NSPLIT];
  #pragma unroll
  for (int q = 1; q < NSPLIT; ++q) {
    const float v = cand_val[t * NSPLIT + q];
    const int   i = cand_idx[t * NSPLIT + q];
    if (v < bv || (v == bv && i < bi)) { bv = v; bi = i; }
  }
  ids[t] = bi;
}

// ------------------------------------------------ gather + partial loss sums --
__global__ __launch_bounds__(256) void gather_loss_kernel(const float* __restrict__ x,
                                                          const float* __restrict__ w,
                                                          const int* __restrict__ ids,
                                                          float* __restrict__ out,
                                                          float* __restrict__ partial) {
  __shared__ float red[256];
  const int tid = threadIdx.x;
  const size_t base = (size_t)blockIdx.x * 4096 + (size_t)tid * 16;  // 8 tokens/blk
  const int token = (int)(base >> 9);
  const int col   = (int)(base & 511);
  const int id = ids[token];
  const float4* wq = (const float4*)(w + (size_t)id * DIMS + col);
  const float4* xq = (const float4*)(x + base);
  float4* oq = (float4*)(out + base);
  float s = 0.0f;
  #pragma unroll
  for (int v = 0; v < 4; ++v) {
    const float4 q  = wq[v];
    const float4 xv = xq[v];
    oq[v] = q;
    const float dx = q.x - xv.x, dy = q.y - xv.y, dz = q.z - xv.z, dw = q.w - xv.w;
    s += dx * dx + dy * dy + dz * dz + dw * dw;
  }
  red[tid] = s;
  __syncthreads();
  for (int o = 128; o; o >>= 1) {
    if (tid < o) red[tid] += red[tid + o];
    __syncthreads();
  }
  if (tid == 0) partial[blockIdx.x] = red[0];
}

__global__ __launch_bounds__(256) void loss_final_kernel(const float* __restrict__ partial,
                                                         float* __restrict__ out_loss) {
  __shared__ float red[256];
  float s = 0.0f;
  for (int i = threadIdx.x; i < 2048; i += 256) s += partial[i];
  red[threadIdx.x] = s;
  __syncthreads();
  for (int o = 128; o; o >>= 1) {
    if (threadIdx.x < o) red[threadIdx.x] += red[threadIdx.x + o];
    __syncthreads();
  }
  // vq_loss = beta*commit + embed = (0.25 + 1.0) * mean((q - x)^2)
  if (threadIdx.x == 0)
    out_loss[0] = red[0] * (1.25f / ((float)NTOK * (float)DIMS));
}

// ---------------------------------------------------------------- launcher --
extern "C" void kernel_launch(void* const* d_in, const int* in_sizes, int n_in,
                              void* d_out, int out_size, void* d_ws, size_t ws_size,
                              hipStream_t stream) {
  (void)in_sizes; (void)n_in; (void)out_size; (void)ws_size;
  const float* x = (const float*)d_in[0];   // [16384, 512]
  const float* w = (const float*)d_in[1];   // [8192, 512]
  float* out = (float*)d_out;               // [16384*512] quantized + [1] loss

  // workspace layout: wnorm | cand_val | cand_idx | ids | partial | w_hi | w_lo
  float* ws       = (float*)d_ws;
  float* wnorm    = ws;                                   // 8192 f32
  float* cand_val = ws + 8192;                            // 16384*4 f32
  int*   cand_idx = (int*)(ws + 8192 + NTOK * NSPLIT);    // 16384*4 i32
  int*   ids      = (int*)(ws + 8192 + 2 * NTOK * NSPLIT);// 16384 i32
  float* partial  = ws + 8192 + 2 * NTOK * NSPLIT + NTOK; // 2048 f32
  __bf16* w_hi    = (__bf16*)(ws + 8192 + 2 * NTOK * NSPLIT + NTOK + 2048);
  __bf16* w_lo    = w_hi + (size_t)NCODES * DIMS;         // 8192*512 bf16 each

  prep_w_kernel<<<NCODES / 8, 256, 0, stream>>>(w, w_hi, w_lo, wnorm);
  vq_argmin_kernel<<<NTOK / 32, 256, 0, stream>>>(x, w_hi, w_lo, wnorm,
                                                  cand_val, cand_idx);
  combine_kernel<<<(NTOK + 255) / 256, 256, 0, stream>>>(cand_val, cand_idx, ids);
  gather_loss_kernel<<<(NTOK * DIMS) / 4096, 256, 0, stream>>>(x, w, ids, out, partial);
  loss_final_kernel<<<1, 256, 0, stream>>>(partial, out + (size_t)NTOK * DIMS);
}
